// ErwinTransolver_69002944578052
// MI455X (gfx1250) — matrix operand order repaired and checked by reading, not compiled, not run on hardware
//
#include <hip/hip_runtime.h>
#include <hip/hip_bf16.h>
#include <math.h>

// ============================================================================
// ErwinTransolver for MI455X (gfx1250, wave32, WMMA).
//
// Pipeline (all f32 storage, bf16 WMMA with f32 accumulate):
//   1. y      = x @ W_inx + b            [65536 x 512, K=256]   WMMA
//   2. temp   = 0.5 + clip(y_bhnc @ w_t) per (b,h,n)            VALU
//   3. L      = y_bhnc @ W_slice + b - log(-log(eps))  (x32)    WMMA
//   4. Wbuf   = transpose(L / temp)  -> (bh, g, n)              LDS transpose
//   5. softmax over n (rows of 16384), in place                 streaming
//   6. es     = Wbuf @ y_bhnc  * 1/(1+1e-5)   (x32, K=16384)    WMMA
//   7. pos    = minmax-normalize es[:, :3]                      reduction
//   8. proc   = erwin ball-tree transformer (32 trees in LDS,
//               tile staged with global_load_async_to_lds)      VALU/LDS/ASYNC
//   9. outp   = proc^T @ Wbuf  -> (b,n,512)   (x32)             WMMA (strided)
//  10. d_out  = outp @ W_out + b             [65536 x 256]      WMMA
// ============================================================================

typedef __attribute__((ext_vector_type(16))) __bf16 v16bf;
typedef __attribute__((ext_vector_type(8)))  float  v8f;

union BF16x16 { v16bf v; unsigned short u[16]; };

__device__ __forceinline__ unsigned short f2bf(float x) {
    unsigned int u = __float_as_uint(x);
    return (unsigned short)((u + 0x7fffu + ((u >> 16) & 1u)) >> 16);
}

// ---------------------------------------------------------------------------
// Generic stride-general batched GEMM.
// One wave computes a 64x16 strip: 4 M-tiles sharing one B fragment per
// K-step (4 independent v_wmma in flight, 4x B-operand reuse).
// C[m,n] = alpha * sum_k A[m,k]*B[k,n] + bias[n] + cadd
// batch offset for operand X: (bi>>3)*xOut + (bi&7)*xIn   (inner batch = 8)
// Requires M%64==0, N%16==0, K%32==0.
// AK1: A is k-contiguous (aks==1) -> A fragment = 4x global_load_b128.
// ---------------------------------------------------------------------------
template<bool AK1>
__global__ void __launch_bounds__(32) gemm_wmma(
    const float* __restrict__ A, const float* __restrict__ B,
    float* __restrict__ C, const float* __restrict__ bias,
    int M, int N, int K,
    long long aOut, long long aIn, long long ams, long long aks,
    long long bOut, long long bIn, long long bks, long long bns,
    long long cOut, long long cIn, long long cms, long long cns,
    float alpha, float cadd)
{
    const int tilesN = N >> 4;
    const int tb = blockIdx.x / tilesN;        // 64-row strip index
    const int tn = blockIdx.x % tilesN;
    const int bi = blockIdx.y;
    const int lane = threadIdx.x & 31;
    const int l15  = lane & 15;
    const int half = lane >> 4;

    const float* Abase = A + (long long)(bi >> 3) * aOut + (long long)(bi & 7) * aIn;
    const float* Bp    = B + (long long)(bi >> 3) * bOut + (long long)(bi & 7) * bIn
                           + (long long)(tn * 16 + l15) * bns;
    const long long m0 = (long long)tb * 64;

    v8f acc[4];
#pragma unroll
    for (int j2 = 0; j2 < 4; ++j2)
        acc[j2] = (v8f){0.f, 0.f, 0.f, 0.f, 0.f, 0.f, 0.f, 0.f};

    for (int kk = 0; kk < K; kk += 32) {
        // B 32x16 bf16 fragment: k = (i>>3)*16 + half*8 + (i&7)   (shared by 4 tiles;
        // coalesced across the half-wave since consecutive lanes hold
        // consecutive n at fixed k)
        BF16x16 bfv;
#pragma unroll
        for (int i = 0; i < 16; ++i) {
            int kb = kk + ((i >> 3) << 4) + (half << 3) + (i & 7);
            bfv.u[i] = f2bf(Bp[(long long)kb * bks]);
        }
        if (kk + 32 < K)
            __builtin_prefetch(Bp + (long long)(kk + 32) * bks, 0, 3);
#pragma unroll
        for (int j2 = 0; j2 < 4; ++j2) {
            // A 16x32 bf16 fragment: k = (i>>3)*16 + half*8 + (i&7)
            const float* Apj = Abase + (m0 + j2 * 16 + l15) * ams;
            BF16x16 af;
            if (AK1) {
                // k-contiguous: whole fragment = 4 x b128 at constant offsets
                const float* base = Apj + kk + (half << 3);
                float4 x0 = *(const float4*)(base + 0);
                float4 x1 = *(const float4*)(base + 4);
                float4 x2 = *(const float4*)(base + 16);
                float4 x3 = *(const float4*)(base + 20);
                af.u[0]  = f2bf(x0.x); af.u[1]  = f2bf(x0.y);
                af.u[2]  = f2bf(x0.z); af.u[3]  = f2bf(x0.w);
                af.u[4]  = f2bf(x1.x); af.u[5]  = f2bf(x1.y);
                af.u[6]  = f2bf(x1.z); af.u[7]  = f2bf(x1.w);
                af.u[8]  = f2bf(x2.x); af.u[9]  = f2bf(x2.y);
                af.u[10] = f2bf(x2.z); af.u[11] = f2bf(x2.w);
                af.u[12] = f2bf(x3.x); af.u[13] = f2bf(x3.y);
                af.u[14] = f2bf(x3.z); af.u[15] = f2bf(x3.w);
            } else {
#pragma unroll
                for (int i = 0; i < 16; ++i) {
                    int ka = kk + ((i >> 3) << 4) + (half << 3) + (i & 7);
                    af.u[i] = f2bf(Apj[(long long)ka * aks]);
                }
            }
            acc[j2] = __builtin_amdgcn_wmma_f32_16x16x32_bf16(
                false, af.v, false, bfv.v, (short)0, acc[j2], false, false);
        }
    }

    const int n = tn * 16 + l15;
    const float bv = bias ? bias[n] : 0.0f;
    float* Cp = C + (long long)(bi >> 3) * cOut + (long long)(bi & 7) * cIn
                  + (long long)n * cns;
#pragma unroll
    for (int j2 = 0; j2 < 4; ++j2)
#pragma unroll
        for (int j = 0; j < 8; ++j) {
            long long m = m0 + j2 * 16 + half * 8 + j;
            Cp[m * cms] = alpha * acc[j2][j] + bv + cadd;
        }
}

// ---------------------------------------------------------------------------
// temp[bh,n] = 0.5 + clip(xp_row . w + b, -0.4, 0.4)
// ---------------------------------------------------------------------------
__global__ void temp_kernel(const float* __restrict__ y,
                            const float* __restrict__ w,
                            const float* __restrict__ b,
                            float* __restrict__ temp, int Npts)
{
    long long i = (long long)blockIdx.x * blockDim.x + threadIdx.x;
    if (i >= (long long)32 * Npts) return;
    int bh = (int)(i / Npts), n = (int)(i % Npts);
    int bb = bh >> 3, h = bh & 7;
    const float* row = y + (long long)bb * Npts * 512 + (long long)n * 512 + h * 64;
    float acc = b[0];
#pragma unroll 8
    for (int d = 0; d < 64; ++d) acc += row[d] * w[d];
    acc = fminf(fmaxf(acc, -0.4f), 0.4f);
    temp[i] = 0.5f + acc;
}

// ---------------------------------------------------------------------------
// W[bh,g,n] = L[bh,n,g] / temp[bh,n]   (64x64 LDS tile transpose)
// ---------------------------------------------------------------------------
__global__ void scale_transpose(const float* __restrict__ L,
                                const float* __restrict__ temp,
                                float* __restrict__ W, int Npts)
{
    __shared__ float tile[64][65];
    __shared__ float it[64];
    const int bh = blockIdx.y;
    const int n0 = blockIdx.x * 64;
    const int tid = threadIdx.x;

    if (tid < 64) it[tid] = 1.0f / temp[(long long)bh * Npts + n0 + tid];
    __syncthreads();
    for (int idx = tid; idx < 4096; idx += 256) {
        int r = idx >> 6, g = idx & 63;
        tile[r][g] = L[((long long)bh * Npts + n0 + r) * 64 + g] * it[r];
    }
    __syncthreads();
    for (int idx = tid; idx < 4096; idx += 256) {
        int g = idx >> 6, r = idx & 63;
        W[((long long)bh * 64 + g) * Npts + n0 + r] = tile[r][g];
    }
}

// ---------------------------------------------------------------------------
// In-place softmax over rows of length Npts (2048 rows). One block per row.
// ---------------------------------------------------------------------------
__global__ void softmax_rows(float* __restrict__ W, int Npts)
{
    __shared__ float red[256];
    const int tid = threadIdx.x;
    float* row = W + (long long)blockIdx.x * Npts;

    float m = -3.402823466e38f;
    for (int i = tid; i < Npts; i += 256) m = fmaxf(m, row[i]);
    red[tid] = m; __syncthreads();
    for (int s = 128; s > 0; s >>= 1) {
        if (tid < s) red[tid] = fmaxf(red[tid], red[tid + s]);
        __syncthreads();
    }
    m = red[0]; __syncthreads();

    float sum = 0.f;
    for (int i = tid; i < Npts; i += 256) sum += expf(row[i] - m);
    red[tid] = sum; __syncthreads();
    for (int s = 128; s > 0; s >>= 1) {
        if (tid < s) red[tid] += red[tid + s];
        __syncthreads();
    }
    const float inv = 1.0f / red[0];
    for (int i = tid; i < Npts; i += 256) row[i] = expf(row[i] - m) * inv;
}

// ---------------------------------------------------------------------------
// pos[r,c] = (es[r,c] - min_c) / (max_c - min_c + 1e-8), c<3, r<2048. 1 block.
// ---------------------------------------------------------------------------
__global__ void pos_kernel(const float* __restrict__ es, float* __restrict__ pos)
{
    __shared__ float rmn[3 * 256], rmx[3 * 256];
    __shared__ float fmn[3], fmx[3];
    const int tid = threadIdx.x;
    float mn[3] = {3.4e38f, 3.4e38f, 3.4e38f};
    float mx[3] = {-3.4e38f, -3.4e38f, -3.4e38f};
    for (int r = tid; r < 2048; r += 256)
        for (int c = 0; c < 3; ++c) {
            float v = es[r * 64 + c];
            mn[c] = fminf(mn[c], v); mx[c] = fmaxf(mx[c], v);
        }
    for (int c = 0; c < 3; ++c) { rmn[c * 256 + tid] = mn[c]; rmx[c * 256 + tid] = mx[c]; }
    __syncthreads();
    for (int s = 128; s > 0; s >>= 1) {
        if (tid < s)
            for (int c = 0; c < 3; ++c) {
                rmn[c * 256 + tid] = fminf(rmn[c * 256 + tid], rmn[c * 256 + tid + s]);
                rmx[c * 256 + tid] = fmaxf(rmx[c * 256 + tid], rmx[c * 256 + tid + s]);
            }
        __syncthreads();
    }
    if (tid < 3) { fmn[tid] = rmn[tid * 256]; fmx[tid] = rmx[tid * 256]; }
    __syncthreads();
    for (int i = tid; i < 2048 * 3; i += 256) {
        int r = i / 3, c = i % 3;
        pos[i] = (es[r * 64 + c] - fmn[c]) / (fmx[c] - fmn[c] + 1e-8f);
    }
}

// ===========================================================================
// Erwin ball-tree transformer: one workgroup (64 threads) per (b,h) tree.
// ===========================================================================
struct BlockPrm {
    const float *ln1g, *ln1b, *ln2g, *ln2b;
    const float *pew, *peb, *projw, *projb, *qkvw, *qkvb, *sigma;
    const float *w1w, *w1b, *w2w, *w2b, *w3w, *w3b;
};
struct ErwinPrm {
    BlockPrm enc0[4], enc1[4], dec0[4];
    const float *embw, *embb;
    const float *poolw, *poolb, *poolg, *poolbeta;
    const float *upw, *upb, *upg, *upbeta;
};
struct Sh { float *A, *B, *Q, *K, *V, *Rel, *Mean; };

__device__ const float ROTM[9] = {
    0.5f, -0.70710678118654752f, 0.5f,
    0.5f,  0.70710678118654752f, 0.5f,
   -0.70710678118654752f, 0.0f,  0.70710678118654752f };

__device__ __forceinline__ void sort_groups(int* ord, const float* coords,
                                            int n, int gsize, int axis)
{
    const int tid = threadIdx.x;
    for (int pass = 0; pass < gsize; ++pass) {
        int start = pass & 1;
        for (int i = tid; i < (n >> 1); i += blockDim.x) {
            int idx = 2 * i + start;
            if (idx + 1 < n && (idx % gsize) != gsize - 1) {
                int a = ord[idx], b = ord[idx + 1];
                if (coords[a * 3 + axis] > coords[b * 3 + axis]) {
                    ord[idx] = b; ord[idx + 1] = a;
                }
            }
        }
        __syncthreads();
    }
}

__device__ __forceinline__ void build_perm_dev(int* out, const float* coords,
                                               int n, int* ord)
{
    const int tid = threadIdx.x;
    for (int i = tid; i < n; i += blockDim.x) ord[i] = i;
    __syncthreads();
    const int levels = (n >= 64) ? 6 : 5;
    for (int l = 0; l < levels; ++l)
        sort_groups(ord, coords, n, n >> l, l % 3);
    for (int i = tid; i < n; i += blockDim.x) out[i] = ord[i];
    __syncthreads();
}

__device__ __forceinline__ void ln_rows(float* dst, const float* src,
                                        const float* g, const float* b,
                                        int n, int dim)
{
    const int t = threadIdx.x;
    if (t < n) {
        const float* r = src + t * dim;
        float mu = 0.f;
        for (int d = 0; d < dim; ++d) mu += r[d];
        mu /= (float)dim;
        float var = 0.f;
        for (int d = 0; d < dim; ++d) { float x = r[d] - mu; var += x * x; }
        var /= (float)dim;
        float rs = rsqrtf(var + 1e-5f);
        float* o = dst + t * dim;
        for (int d = 0; d < dim; ++d) o[d] = (r[d] - mu) * rs * g[d] + b[d];
    }
    __syncthreads();
}

__device__ __forceinline__ void ball_rel(const float* p, float* rel, float* means,
                                         int n, int ball)
{
    const int t = threadIdx.x;
    const int nb = n / ball;
    if (t < nb * 3) {
        int bi = t / 3, c = t % 3;
        float s = 0.f;
        for (int j = 0; j < ball; ++j) s += p[(bi * ball + j) * 3 + c];
        means[t] = s / (float)ball;
    }
    __syncthreads();
    if (t < n) {
        int bi = t / ball;
        for (int c = 0; c < 3; ++c) rel[t * 3 + c] = p[t * 3 + c] - means[bi * 3 + c];
    }
    __syncthreads();
}

template<int N_, int DIM, int BALL>
__device__ void tf_block(float* f, const float* p, const BlockPrm& pr, Sh sh)
{
    const int HD = DIM / 8;
    const int t = threadIdx.x;

    // --- ln1, pe(rel) ---
    ln_rows(sh.A, f, pr.ln1g, pr.ln1b, N_, DIM);
    ball_rel(p, sh.Rel, sh.Mean, N_, BALL);
    if (t < N_) {
        float r0 = sh.Rel[t * 3], r1 = sh.Rel[t * 3 + 1], r2 = sh.Rel[t * 3 + 2];
        for (int d = 0; d < DIM; ++d)
            sh.A[t * DIM + d] += pr.peb[d] + r0 * pr.pew[d]
                               + r1 * pr.pew[DIM + d] + r2 * pr.pew[2 * DIM + d];
    }
    __syncthreads();

    // --- attention, head by head (q/k/v tiles live in LDS) ---
    const float isq = rsqrtf((float)HD);
    for (int hh = 0; hh < 8; ++hh) {
        if (t < N_) {
            for (int d = 0; d < HD; ++d) {
                float q = pr.qkvb[hh * HD + d];
                float k = pr.qkvb[DIM + hh * HD + d];
                float v = pr.qkvb[2 * DIM + hh * HD + d];
                for (int c = 0; c < DIM; ++c) {
                    float a = sh.A[t * DIM + c];
                    const float* wrow = pr.qkvw + (long long)c * 3 * DIM;
                    q += a * wrow[hh * HD + d];
                    k += a * wrow[DIM + hh * HD + d];
                    v += a * wrow[2 * DIM + hh * HD + d];
                }
                sh.Q[t * HD + d] = q; sh.K[t * HD + d] = k; sh.V[t * HD + d] = v;
            }
        }
        __syncthreads();
        if (t < N_) {
            const int base = (t / BALL) * BALL;
            const float sg = pr.sigma[hh];
            float lg[BALL];
            float mx = -3.4e38f;
            for (int j = 0; j < BALL; ++j) {
                int jj = base + j;
                float dot = 0.f;
#pragma unroll
                for (int d = 0; d < HD; ++d) dot += sh.Q[t * HD + d] * sh.K[jj * HD + d];
                float dx = sh.Rel[t * 3]     - sh.Rel[jj * 3];
                float dy = sh.Rel[t * 3 + 1] - sh.Rel[jj * 3 + 1];
                float dz = sh.Rel[t * 3 + 2] - sh.Rel[jj * 3 + 2];
                float l = dot * isq + sg * sqrtf(dx * dx + dy * dy + dz * dz);
                lg[j] = l; mx = fmaxf(mx, l);
            }
            float s = 0.f;
            for (int j = 0; j < BALL; ++j) { lg[j] = expf(lg[j] - mx); s += lg[j]; }
            const float inv = 1.0f / s;
            float acc[HD];
#pragma unroll
            for (int d = 0; d < HD; ++d) acc[d] = 0.f;
            for (int j = 0; j < BALL; ++j) {
                float w = lg[j] * inv;
#pragma unroll
                for (int d = 0; d < HD; ++d) acc[d] += w * sh.V[(base + j) * HD + d];
            }
#pragma unroll
            for (int d = 0; d < HD; ++d) sh.B[t * DIM + hh * HD + d] = acc[d];
        }
        __syncthreads();
    }

    // --- proj + residual ---
    if (t < N_) {
        for (int d = 0; d < DIM; ++d) {
            float a = pr.projb[d];
            for (int c = 0; c < DIM; ++c) a += sh.B[t * DIM + c] * pr.projw[c * DIM + d];
            f[t * DIM + d] += a;
        }
    }
    __syncthreads();

    // --- MLP: f += (silu(h@w1) * (h@w2)) @ w3 ---
    ln_rows(sh.A, f, pr.ln2g, pr.ln2b, N_, DIM);
    if (t < N_) {
        for (int d = 0; d < DIM; ++d) sh.B[t * DIM + d] = 0.f;
        for (int j = 0; j < 4 * DIM; ++j) {
            float u = pr.w1b[j], w = pr.w2b[j];
            for (int c = 0; c < DIM; ++c) {
                float a = sh.A[t * DIM + c];
                u += a * pr.w1w[c * 4 * DIM + j];
                w += a * pr.w2w[c * 4 * DIM + j];
            }
            float sv = (u / (1.0f + expf(-u))) * w;
            for (int d = 0; d < DIM; ++d) sh.B[t * DIM + d] += sv * pr.w3w[j * DIM + d];
        }
        for (int d = 0; d < DIM; ++d) f[t * DIM + d] += sh.B[t * DIM + d] + pr.w3b[d];
    }
    __syncthreads();
}

template<int N_, int DIM, int BALL>
__device__ void run_blocks_dev(float* sF, float* sG, const float* p, float* sPp,
                               const BlockPrm* blocks, const int* rot, Sh sh)
{
    const int t = threadIdx.x;
    for (int i = 0; i < 4; ++i) {
        if (i & 1) {
            if (t < N_) {
                int s = rot[t];
                for (int d = 0; d < DIM; ++d) sG[t * DIM + d] = sF[s * DIM + d];
                for (int c = 0; c < 3; ++c)  sPp[t * 3 + c] = p[s * 3 + c];
            }
            __syncthreads();
            tf_block<N_, DIM, BALL>(sG, sPp, blocks[i], sh);
            if (t < N_) {
                int s = rot[t];
                for (int d = 0; d < DIM; ++d) sF[s * DIM + d] = sG[t * DIM + d];
            }
            __syncthreads();
        } else {
            tf_block<N_, DIM, BALL>(sF, p, blocks[i], sh);
        }
    }
}

__global__ void __launch_bounds__(64) erwin_kernel(
    const float* __restrict__ es, const float* __restrict__ pos,
    float* __restrict__ skipWS, float* __restrict__ proc, ErwinPrm ep)
{
    extern __shared__ float smem[];
    float* sF   = smem;          // 4096
    float* sG   = sF + 4096;     // 4096
    float* sA   = sG + 4096;     // 4096
    float* sB   = sA + 4096;     // 4096
    float* sQ   = sB + 4096;     // 1024
    float* sK   = sQ + 1024;     // 1024
    float* sV   = sK + 1024;     // 1024
    float* sP0  = sV + 1024;     // 192
    float* sPR  = sP0 + 192;     // 192
    float* sPp  = sPR + 192;     // 192
    float* sP1  = sPp + 192;     // 96
    float* sRel = sP1 + 96;      // 192
    float* sMean = sRel + 192;   // 8
    int* sPerm = (int*)(sMean + 8);  // 64
    int* sRot0 = sPerm + 64;         // 64
    int* sRot1 = sRot0 + 64;         // 32
    int* sOrd  = sRot1 + 32;         // 64

    Sh sh = { sA, sB, sQ, sK, sV, sRel, sMean };
    const int t = threadIdx.x;
    const int bh = blockIdx.x;

    // ---- stage the tree's 64x64 feature tile into LDS (sG) with the CDNA5
    // async DMA path, overlapped with building the kd-tree permutation ----
    {
        unsigned lds_base = (unsigned)(unsigned long long)(void*)sG;
        const float* gsrc = es + (long long)bh * 4096;
        for (int i = t; i < 1024; i += 64) {            // 1024 x 16B = 16 KB
            unsigned lds_addr = lds_base + (unsigned)(i * 16);
            unsigned long long gaddr = (unsigned long long)(gsrc + i * 4);
            asm volatile("global_load_async_to_lds_b128 %0, %1, off"
                         :: "v"(lds_addr), "v"(gaddr));
        }
    }

    // original positions -> sPR, build main perm (DMA in flight meanwhile)
    if (t < 64)
        for (int c = 0; c < 3; ++c) sPR[t * 3 + c] = pos[(bh * 64 + t) * 3 + c];
    __syncthreads();
    build_perm_dev(sPerm, sPR, 64, sOrd);

    // wait for the async tile, then permute features + positions
    asm volatile("s_wait_asynccnt 0x0" ::: "memory");
    __syncthreads();
    if (t < 64) {
        int s = sPerm[t];
        for (int d = 0; d < 64; ++d) sF[t * 64 + d] = sG[s * 64 + d];
        for (int c = 0; c < 3; ++c) sP0[t * 3 + c] = sPR[s * 3 + c];
    }
    __syncthreads();

    // rot0 = build_perm(p @ ROT^T)
    if (t < 64)
        for (int c = 0; c < 3; ++c)
            sPR[t * 3 + c] = ROTM[c * 3] * sP0[t * 3] + ROTM[c * 3 + 1] * sP0[t * 3 + 1]
                           + ROTM[c * 3 + 2] * sP0[t * 3 + 2];
    __syncthreads();
    build_perm_dev(sRot0, sPR, 64, sOrd);

    // embedding 64->64
    if (t < 64)
        for (int d = 0; d < 64; ++d) {
            float a = ep.embb[d];
            for (int k = 0; k < 64; ++k) a += sF[t * 64 + k] * ep.embw[k * 64 + d];
            sG[t * 64 + d] = a;
        }
    __syncthreads();
    if (t < 64) for (int d = 0; d < 64; ++d) sF[t * 64 + d] = sG[t * 64 + d];
    __syncthreads();

    // encoder 0 (dim 64, ball 32)
    run_blocks_dev<64, 64, 32>(sF, sG, sP0, sPp, ep.enc0, sRot0, sh);

    // spill skip connection to global scratch
    if (t < 64)
        for (int d = 0; d < 64; ++d)
            skipWS[(long long)bh * 4096 + t * 64 + d] = sF[t * 64 + d];
    __syncthreads();

    // ---- pool stride 2 ----
    if (t < 32)
        for (int c = 0; c < 3; ++c)
            sP1[t * 3 + c] = 0.5f * (sP0[(2 * t) * 3 + c] + sP0[(2 * t + 1) * 3 + c]);
    __syncthreads();
    if (t < 32) {
        for (int d = 0; d < 128; ++d) {
            float a = ep.poolb[d];
            for (int k = 0; k < 128; ++k) a += sF[t * 128 + k] * ep.poolw[k * 128 + d];
            for (int k = 0; k < 6; ++k) {
                int pair = k / 3, c = k % 3;
                float rv = sP0[(2 * t + pair) * 3 + c] - sP1[t * 3 + c];
                a += rv * ep.poolw[(128 + k) * 128 + d];
            }
            sA[t * 128 + d] = a;
        }
    }
    __syncthreads();
    ln_rows(sF, sA, ep.poolg, ep.poolbeta, 32, 128);

    // rot1 = build_perm(p1 @ ROT^T), n=32
    if (t < 32)
        for (int c = 0; c < 3; ++c)
            sPR[t * 3 + c] = ROTM[c * 3] * sP1[t * 3] + ROTM[c * 3 + 1] * sP1[t * 3 + 1]
                           + ROTM[c * 3 + 2] * sP1[t * 3 + 2];
    __syncthreads();
    build_perm_dev(sRot1, sPR, 32, sOrd);

    // encoder 1 (dim 128, ball 16, n=32)
    run_blocks_dev<32, 128, 16>(sF, sG, sP1, sPp, ep.enc1, sRot1, sh);

    // ---- unpool + skip ----
    if (t < 32) {
        for (int d = 0; d < 128; ++d) {
            float a = ep.upb[d];
            for (int k = 0; k < 128; ++k) a += sF[t * 128 + k] * ep.upw[k * 128 + d];
            for (int k = 0; k < 6; ++k) {
                int pair = k / 3, c = k % 3;
                float rv = sP0[(2 * t + pair) * 3 + c] - sP1[t * 3 + c];
                a += rv * ep.upw[(128 + k) * 128 + d];
            }
            sA[t * 128 + d] = a;
        }
    }
    __syncthreads();
    if (t < 64)
        for (int d = 0; d < 64; ++d)
            sG[t * 64 + d] = skipWS[(long long)bh * 4096 + t * 64 + d] + sA[t * 64 + d];
    __syncthreads();
    ln_rows(sF, sG, ep.upg, ep.upbeta, 64, 64);

    // decoder (dim 64, ball 32)
    run_blocks_dev<64, 64, 32>(sF, sG, sP0, sPp, ep.dec0, sRot0, sh);

    // un-permute and write out: out[perm[t]] = f[t]
    if (t < 64) {
        int g = sPerm[t];
        for (int d = 0; d < 64; ++d)
            proc[(long long)bh * 4096 + g * 64 + d] = sF[t * 64 + d];
    }
}

// ===========================================================================
// Host launcher
// ===========================================================================
static BlockPrm mk_block(void* const* d_in, int base)
{
    auto f = [&](int i) { return (const float*)d_in[i]; };
    BlockPrm p;
    p.ln1b = f(base + 0);  p.ln1g = f(base + 1);
    p.ln2b = f(base + 2);  p.ln2g = f(base + 3);
    p.peb  = f(base + 4);  p.pew  = f(base + 5);
    p.projb = f(base + 6); p.projw = f(base + 7);
    p.qkvb = f(base + 8);  p.qkvw = f(base + 9);
    p.sigma = f(base + 10);
    p.w1b = f(base + 11);  p.w1w = f(base + 12);
    p.w2b = f(base + 13);  p.w2w = f(base + 14);
    p.w3b = f(base + 15);  p.w3w = f(base + 16);
    return p;
}

extern "C" void kernel_launch(void* const* d_in, const int* in_sizes, int n_in,
                              void* d_out, int out_size, void* d_ws, size_t ws_size,
                              hipStream_t stream)
{
    // Input order assumption: d_in[0] = x; d_in[1..222] = params leaves,
    // jax tree order (sorted dict keys): dec0[0..3], emb, enc0[0..3],
    // enc1[0..3], in_x, out, pool0(ln,proj), slice, temp, unpool0(ln,proj);
    // each _lin -> (b,w), each _ln -> (b,g), each block -> 17 leaves.
    if (n_in < 223) return;
    const int Npts = 16384;
    auto f = [&](int i) { return (const float*)d_in[i]; };
    const float* x = f(0);

    // param indices (base P=1)
    const int DEC0 = 1, EMB_B = 69, EMB_W = 70, ENC0 = 71, ENC1 = 139;
    const int INX_B = 207, INX_W = 208, OUT_B = 209, OUT_W = 210;
    const int PL_LNB = 211, PL_LNG = 212, PL_PB = 213, PL_PW = 214;
    const int SL_B = 215, SL_W = 216, TP_B = 217, TP_W = 218;
    const int UP_LNB = 219, UP_LNG = 220, UP_PB = 221, UP_PW = 222;

    // workspace carve (floats)
    float* ws = (float*)d_ws;
    const long long SZy = 33554432LL;      // (B,N,512)
    float* y    = ws;
    float* Wbuf = y + SZy;                 // (bh,g,n)  33554432
    float* Lbuf = Wbuf + SZy;              // logits (bh,n,g), reused as out_pre
    float* temp = Lbuf + SZy;              // 524288
    float* es   = temp + 524288;           // 131072
    float* proc = es + 131072;             // 131072
    float* pos  = proc + 131072;           // 8192
    float* skip = pos + 8192;              // 131072
    if (ws_size < (size_t)(3 * SZy + 524288 + 3 * 131072 + 8192) * sizeof(float)) return;

    // 1. y = x @ W_inx + b   [65536 x 512, K=256]   (A k-contiguous)
    {
        dim3 grid((65536 / 64) * (512 / 16), 1);
        gemm_wmma<true><<<grid, 32, 0, stream>>>(x, f(INX_W), y, f(INX_B),
            65536, 512, 256,
            0, 0, 256, 1,   0, 0, 512, 1,   0, 0, 512, 1,
            1.0f, 0.0f);
    }
    // 2. temp
    temp_kernel<<<(32 * Npts) / 256, 256, 0, stream>>>(y, f(TP_W), f(TP_B), temp, Npts);
    // 3. logits (batched over 32 bh): L[bh,n,g]   (A k-contiguous)
    {
        dim3 grid((Npts / 64) * (64 / 16), 32);
        const float cadd = -logf(-logf(1e-6f));
        gemm_wmma<true><<<grid, 32, 0, stream>>>(y, f(SL_W), Lbuf, f(SL_B),
            Npts, 64, 64,
            (long long)Npts * 512, 64, 512, 1,
            0, 0, 64, 1,
            8LL * Npts * 64, (long long)Npts * 64, 64, 1,
            1.0f, cadd);
    }
    // 4. Wbuf[bh,g,n] = L/temp transposed
    {
        dim3 grid(Npts / 64, 32);
        scale_transpose<<<grid, 256, 0, stream>>>(Lbuf, temp, Wbuf, Npts);
    }
    // 5. softmax over n, in place (2048 rows)
    softmax_rows<<<2048, 256, 0, stream>>>(Wbuf, Npts);
    // 6. es[bh,g,c] = (1/(1+1e-5)) * Wbuf[bh,g,:] @ y[bh,:,c]   K=16384
    {
        dim3 grid((64 / 64) * (64 / 16), 32);
        gemm_wmma<true><<<grid, 32, 0, stream>>>(Wbuf, y, es, nullptr,
            64, 64, Npts,
            8LL * 64 * Npts, 64LL * Npts, Npts, 1,
            (long long)Npts * 512, 64, 512, 1,
            8LL * 4096, 4096, 64, 1,
            1.0f / (1.0f + 1e-5f), 0.0f);
    }
    // 7. positions
    pos_kernel<<<1, 256, 0, stream>>>(es, pos);
    // 8. erwin (32 trees)
    {
        ErwinPrm ep;
        for (int i = 0; i < 4; ++i) {
            ep.dec0[i] = mk_block(d_in, DEC0 + 17 * i);
            ep.enc0[i] = mk_block(d_in, ENC0 + 17 * i);
            ep.enc1[i] = mk_block(d_in, ENC1 + 17 * i);
        }
        ep.embw = f(EMB_W);   ep.embb = f(EMB_B);
        ep.poolw = f(PL_PW);  ep.poolb = f(PL_PB);
        ep.poolg = f(PL_LNG); ep.poolbeta = f(PL_LNB);
        ep.upw = f(UP_PW);    ep.upb = f(UP_PB);
        ep.upg = f(UP_LNG);   ep.upbeta = f(UP_LNB);
        const size_t shmem = (4 * 4096 + 3 * 1024 + 192 * 4 + 96 + 8) * sizeof(float)
                           + (64 + 64 + 32 + 64) * sizeof(int);
        erwin_kernel<<<32, 64, shmem, stream>>>(es, pos, skip, proc, ep);
    }
    // 9. out_pre[b,n,h*64+c] = sum_g proc[bh,g,c] * Wbuf[bh,g,n]  (M=c,N=n,K=g)
    //    A = proc^T via strides (aks=64) -> general path
    {
        dim3 grid((64 / 64) * (Npts / 16), 32);
        gemm_wmma<false><<<grid, 32, 0, stream>>>(proc, Wbuf, Lbuf, nullptr,
            64, Npts, 64,
            8LL * 4096, 4096, 1, 64,
            8LL * 64 * Npts, 64LL * Npts, Npts, 1,
            (long long)Npts * 512, 64, 1, 512,
            1.0f, 0.0f);
    }
    // 10. d_out = out_pre @ W_out + b   [65536 x 256, K=512]   (A k-contiguous)
    {
        dim3 grid((65536 / 64) * (256 / 16), 1);
        gemm_wmma<true><<<grid, 32, 0, stream>>>(Lbuf, f(OUT_W), (float*)d_out, f(OUT_B),
            65536, 256, 512,
            0, 0, 512, 1,   0, 0, 256, 1,   0, 0, 256, 1,
            1.0f, 0.0f);
    }
}
